// ConvJointNet_66511863545984
// MI455X (gfx1250) — matrix operand order repaired and checked by reading, not compiled, not run on hardware
//
#include <hip/hip_runtime.h>

// ---------------------------------------------------------------------------
// Types for CDNA5 WMMA (wave32): v16h A/B operands, v8f accumulator.
// ---------------------------------------------------------------------------
typedef _Float16 half8 __attribute__((ext_vector_type(8)));
typedef _Float16 v16h  __attribute__((ext_vector_type(16)));
typedef float    v8f   __attribute__((ext_vector_type(8)));

union V16H { v16h v; half8 h[2]; };

// Problem constants (from reference)
#define NB   8
#define TT   200
#define UU   50
#define EE   512
#define KK   512     // joint dim
#define CC   1024    // classes
#define MENC (NB*TT)     // 1600
#define MDEC (NB*UU)     // 400
#define MTU  (NB*TT*UU)  // 80000

// ---------------------------------------------------------------------------
// Async-stage a 16x512 f16 A-panel (16 KB) from global into LDS.
// 256 threads x 4 chunks x 16 B = 16 KB, via global_load_async_to_lds_b128
// (ASYNCcnt path). The LDS VGPR operand is the LDS byte address = low 32 bits
// of the flat shared-aperture address.
// Caller must __syncthreads() after this before consuming the panel.
// ---------------------------------------------------------------------------
__device__ inline void async_stage_panel_16x512(const _Float16* __restrict__ g,
                                                _Float16* lds, int tid) {
    uint32_t lbase = (uint32_t)(uintptr_t)lds;
    uint64_t gbase = (uint64_t)(uintptr_t)g;
    #pragma unroll
    for (int i = 0; i < 4; ++i) {
        uint32_t idx  = (uint32_t)tid + i * 256;       // 16B chunk id, 0..1023
        uint32_t loff = lbase + idx * 16;
        uint64_t ga   = gbase + (uint64_t)idx * 16;
        asm volatile("global_load_async_to_lds_b128 %0, %1, off"
                     :: "v"(loff), "v"(ga) : "memory");
    }
    asm volatile("s_wait_asynccnt 0" ::: "memory");
}

// ---------------------------------------------------------------------------
// A-tile loader from the LDS panel: 16x32 f16 tile at column k0.
// ISA layout (16-bit A 16x32): lanes 0-15 hold row M=lane, K = k0+[0..7] in
// v16h elems 0..7 and K = k0+16+[0..7] in elems 8..15; lanes 16-31 same rows,
// K offset by +8.
// ---------------------------------------------------------------------------
__device__ inline v16h load_a_tile_lds(const _Float16* aPanel, int k0, int lane) {
    int sel = lane >> 4;
    int row = lane & 15;
    const _Float16* p = aPanel + row * KK + k0 + sel * 8;
    V16H r;
    r.h[0] = *(const half8*)(p);
    r.h[1] = *(const half8*)(p + 16);
    return r.v;
}

// ---------------------------------------------------------------------------
// B-tile loader: 32x16 f16 (K x N) taken from row-major weight W[n][k]
// (so B[k][n] = W[n][k], exactly the transpose the contraction needs).
// ISA layout: lane = N, v16h elems 0..7 hold K = k0+8*sel+[0..7] and
// elems 8..15 hold K = k0+16+8*sel+[0..7] (same order as the A tile).
// ---------------------------------------------------------------------------
__device__ inline v16h load_b_tile(const _Float16* __restrict__ W, int ldw,
                                   int nBase, int k0, int lane) {
    int sel = lane >> 4;
    int col = nBase + (lane & 15);
    const _Float16* p = W + (size_t)col * ldw + k0 + sel * 8;
    V16H r;
    r.h[0] = *(const half8*)(p);
    r.h[1] = *(const half8*)(p + 16);
    return r.v;
}

// ---------------------------------------------------------------------------
// Kernel 0: f32 -> f16 pack (inputs + weights into workspace)
// ---------------------------------------------------------------------------
__global__ __launch_bounds__(256) void f32_to_f16_kernel(
        const float* __restrict__ src, _Float16* __restrict__ dst, int n) {
    int i = blockIdx.x * 256 + threadIdx.x;
    if (i < n) dst[i] = (_Float16)src[i];
}

// ---------------------------------------------------------------------------
// Kernel 1: projection GEMM  out[m][n] = sum_e A[m][e] * W[n][e] + bias[n]
// M = gridDim.x*16, N = K = 512. 256 threads = 8 waves, each wave owns
// 4 N-tiles of 16 -> full 512 columns per block. A panel staged via async.
// ---------------------------------------------------------------------------
__global__ __launch_bounds__(256) void proj_gemm_kernel(
        const _Float16* __restrict__ A, const _Float16* __restrict__ W,
        const float* __restrict__ bias, float* __restrict__ out) {
    __shared__ _Float16 aPanel[16 * EE];   // 16 KB

    const int mBase = blockIdx.x * 16;
    const int lane  = threadIdx.x & 31;
    const int wave  = threadIdx.x >> 5;

    async_stage_panel_16x512(A + (size_t)mBase * EE, aPanel, threadIdx.x);
    __syncthreads();

    v8f acc[4];
    #pragma unroll
    for (int j = 0; j < 4; ++j)
        #pragma unroll
        for (int i = 0; i < 8; ++i) acc[j][i] = 0.0f;

    for (int k0 = 0; k0 < EE; k0 += 32) {
        v16h a = load_a_tile_lds(aPanel, k0, lane);
        #pragma unroll
        for (int j = 0; j < 4; ++j) {
            int nBase = (wave * 4 + j) * 16;
            v16h b = load_b_tile(W, EE, nBase, k0, lane);
            acc[j] = __builtin_amdgcn_wmma_f32_16x16x32_f16(
                false, a, false, b, (short)0, acc[j], false, false);
        }
    }

    const int sel = lane >> 4, ncol = lane & 15;
    #pragma unroll
    for (int j = 0; j < 4; ++j) {
        int nBase = (wave * 4 + j) * 16;
        float bv = bias[nBase + ncol];
        #pragma unroll
        for (int v = 0; v < 8; ++v) {
            int m = mBase + v + 8 * sel;
            out[(size_t)m * KK + nBase + ncol] = acc[j][v] + bv;
        }
    }
}

// ---------------------------------------------------------------------------
// Kernel 2: fused tanh(enc+dec) + causal 3x3 depthwise conv + f16 pack.
// One block per (n,t,u); 512 threads over channels k. Padding is applied
// AFTER tanh in the reference, so out-of-range taps contribute exactly 0
// and must be skipped (tanh(enc+dec) != 0 in general).
// ---------------------------------------------------------------------------
__global__ __launch_bounds__(512) void act_dwconv_kernel(
        const float* __restrict__ enc, const float* __restrict__ dec,
        const float* __restrict__ dw,  const float* __restrict__ db,
        _Float16* __restrict__ Y) {
    const int m = blockIdx.x;           // 0 .. MTU-1
    const int k = threadIdx.x;          // 0 .. 511
    const int u = m % UU;
    const int t = (m / UU) % TT;
    const int n = m / (UU * TT);

    float ev[3], dv[3];
    bool  eok[3], dok[3];
    #pragma unroll
    for (int d = 0; d < 3; ++d) {
        int tt = t - 2 + d;
        eok[d] = (tt >= 0);
        ev[d]  = eok[d] ? enc[((size_t)(n * TT + tt)) * KK + k] : 0.0f;
        int uu = u - 2 + d;
        dok[d] = (uu >= 0);
        dv[d]  = dok[d] ? dec[((size_t)(n * UU + uu)) * KK + k] : 0.0f;
    }

    float acc = db[k];
    #pragma unroll
    for (int dt = 0; dt < 3; ++dt)
        #pragma unroll
        for (int du = 0; du < 3; ++du)
            if (eok[dt] && dok[du])
                acc += dw[(size_t)k * 9 + dt * 3 + du] * tanhf(ev[dt] + dv[du]);

    Y[(size_t)m * KK + k] = (_Float16)acc;
}

// ---------------------------------------------------------------------------
// Kernel 3: pointwise GEMM (80000x1024x512) + bias + fused log_softmax.
// 16-row M-tile per block; 8 waves x 8 N-tiles of 16 cover all 1024 classes
// so the softmax reduction never leaves the workgroup. The A panel (16 KB) is
// async-staged into LDS once and reused by all waves/K-steps; B streams from
// the L2-resident 1 MB f16 point_w. GEMM result parks in a 64 KB LDS tile,
// then max / log-sum-exp across classes and a coalesced 328 MB output write.
// ---------------------------------------------------------------------------
__global__ __launch_bounds__(256) void pointwise_lsm_kernel(
        const _Float16* __restrict__ Y, const _Float16* __restrict__ PW,
        const float* __restrict__ pb, float* __restrict__ out) {
    __shared__ float    zbuf[16][CC];    // 64 KB
    __shared__ _Float16 aPanel[16 * KK]; // 16 KB
    __shared__ float    pred[16][16];
    __shared__ float    rmax[16];
    __shared__ float    rlse[16];

    const int mBase = blockIdx.x * 16;
    const int lane  = threadIdx.x & 31;
    const int wave  = threadIdx.x >> 5;

    async_stage_panel_16x512(Y + (size_t)mBase * KK, aPanel, threadIdx.x);
    __syncthreads();

    v8f acc[8];
    #pragma unroll
    for (int j = 0; j < 8; ++j)
        #pragma unroll
        for (int i = 0; i < 8; ++i) acc[j][i] = 0.0f;

    for (int k0 = 0; k0 < KK; k0 += 32) {
        v16h a = load_a_tile_lds(aPanel, k0, lane);
        if (k0 + 32 < KK) {  // keep next K-step's B slab warm (global_prefetch_b8)
            const _Float16* pf = PW + (size_t)((wave * 8) * 16 + (lane & 15)) * KK
                                    + (k0 + 32) + (lane >> 4) * 16;
            __builtin_prefetch(pf, 0, 0);
        }
        #pragma unroll
        for (int j = 0; j < 8; ++j) {
            int nBase = (wave * 8 + j) * 16;
            v16h b = load_b_tile(PW, KK, nBase, k0, lane);
            acc[j] = __builtin_amdgcn_wmma_f32_16x16x32_f16(
                false, a, false, b, (short)0, acc[j], false, false);
        }
    }

    // C layout: lane holds N = lane&15 (nBase+), rows M = v + 8*(lane>>4).
    const int sel = lane >> 4, ncol = lane & 15;
    #pragma unroll
    for (int j = 0; j < 8; ++j) {
        int nBase = (wave * 8 + j) * 16;
        float bv = pb[nBase + ncol];
        #pragma unroll
        for (int v = 0; v < 8; ++v)
            zbuf[v + 8 * sel][nBase + ncol] = acc[j][v] + bv;
    }
    __syncthreads();

    // log_softmax over 1024 classes: 16 threads per row.
    const int row = threadIdx.x >> 4;
    const int cgr = threadIdx.x & 15;

    float mx = -INFINITY;
    for (int c = cgr; c < CC; c += 16) mx = fmaxf(mx, zbuf[row][c]);
    pred[row][cgr] = mx;
    __syncthreads();
    if (cgr == 0) {
        float m2 = -INFINITY;
        #pragma unroll
        for (int i = 0; i < 16; ++i) m2 = fmaxf(m2, pred[row][i]);
        rmax[row] = m2;
    }
    __syncthreads();

    const float rm = rmax[row];
    float s = 0.0f;
    for (int c = cgr; c < CC; c += 16) s += expf(zbuf[row][c] - rm);
    __syncthreads();            // pred reuse hazard
    pred[row][cgr] = s;
    __syncthreads();
    if (cgr == 0) {
        float s2 = 0.0f;
        #pragma unroll
        for (int i = 0; i < 16; ++i) s2 += pred[row][i];
        rlse[row] = logf(s2);
    }
    __syncthreads();

    for (int idx = threadIdx.x; idx < 16 * CC; idx += 256) {
        int r = idx >> 10, c = idx & (CC - 1);
        out[(size_t)(mBase + r) * CC + c] = zbuf[r][c] - rmax[r] - rlse[r];
    }
}

// ---------------------------------------------------------------------------
// Host launcher
// ---------------------------------------------------------------------------
extern "C" void kernel_launch(void* const* d_in, const int* in_sizes, int n_in,
                              void* d_out, int out_size, void* d_ws, size_t ws_size,
                              hipStream_t stream) {
    (void)in_sizes; (void)n_in; (void)out_size; (void)ws_size;

    const float* enc_in = (const float*)d_in[0];  // [8,200,512]
    const float* dec_in = (const float*)d_in[1];  // [8,50,512]
    const float* W_enc  = (const float*)d_in[2];  // [512,512]
    const float* b_enc  = (const float*)d_in[3];  // [512]
    const float* W_dec  = (const float*)d_in[4];  // [512,512]
    const float* b_dec  = (const float*)d_in[5];  // [512]
    const float* dw     = (const float*)d_in[6];  // [512,1,3,3]
    const float* db     = (const float*)d_in[7];  // [512]
    const float* pw     = (const float*)d_in[8];  // [1024,512,1,1]
    const float* pb     = (const float*)d_in[9];  // [1024]
    float* out = (float*)d_out;                   // [8,200,50,1024]

    // Workspace layout (256B aligned segments)
    char* ws = (char*)d_ws;
    size_t off = 0;
    auto take = [&](size_t bytes) {
        size_t o = off;
        off = (off + bytes + 255) & ~(size_t)255;
        return o;
    };
    _Float16* encH  = (_Float16*)(ws + take((size_t)MENC * EE * 2)); // 1.64 MB
    _Float16* decH  = (_Float16*)(ws + take((size_t)MDEC * EE * 2));
    _Float16* WencH = (_Float16*)(ws + take((size_t)KK * EE * 2));
    _Float16* WdecH = (_Float16*)(ws + take((size_t)KK * EE * 2));
    _Float16* pwH   = (_Float16*)(ws + take((size_t)CC * KK * 2));   // 1 MB, lives in L2
    float*    encP  = (float*)(ws + take((size_t)MENC * KK * 4));    // projected enc
    float*    decP  = (float*)(ws + take((size_t)MDEC * KK * 4));    // projected dec
    _Float16* yH    = (_Float16*)(ws + take((size_t)MTU * KK * 2));  // 82 MB conv out

    // 0) f32 -> f16 packs
    auto cvt = [&](const float* s, _Float16* d, int n) {
        f32_to_f16_kernel<<<(n + 255) / 256, 256, 0, stream>>>(s, d, n);
    };
    cvt(enc_in, encH, MENC * EE);
    cvt(dec_in, decH, MDEC * EE);
    cvt(W_enc,  WencH, KK * EE);
    cvt(W_dec,  WdecH, KK * EE);
    cvt(pw,     pwH,   CC * KK);

    // 1) projections (WMMA, async-staged A panels)
    proj_gemm_kernel<<<MENC / 16, 256, 0, stream>>>(encH, WencH, b_enc, encP);
    proj_gemm_kernel<<<MDEC / 16, 256, 0, stream>>>(decH, WdecH, b_dec, decP);

    // 2) tanh + causal depthwise conv, packed f16
    act_dwconv_kernel<<<MTU, 512, 0, stream>>>(encP, decP, dw, db, yH);

    // 3) pointwise GEMM + bias + log_softmax (WMMA, fused epilogue)
    pointwise_lsm_kernel<<<MTU / 16, 256, 0, stream>>>(yH, pwH, pb, out);
}
